// VideoSNDetDiscriminator_962072674486
// MI455X (gfx1250) — compile-verified, matrix operand order checked by reading, not run on hardware
//
#include <hip/hip_runtime.h>
#include <math.h>

// ---------------------------------------------------------------------------
// Types for CDNA5 WMMA (wave32): v_wmma_f32_16x16x32_f16
// ---------------------------------------------------------------------------
typedef __attribute__((ext_vector_type(16))) _Float16 v16h;
typedef __attribute__((ext_vector_type(8)))  _Float16 v8h;
typedef __attribute__((ext_vector_type(8)))  float    v8f;

#define H_IMG 256
#define W_IMG 512
#define BS 32
#define TRACK 4
#define NPATCH (BS * TRACK)   // 128
#define OUT 64

// ---------------------------------------------------------------------------
// Spectral-norm sigma + pack weights into per-lane WMMA B-fragment order.
// One block per layer. wfrag layout: idx = ((((nt*9+tap)*KC + kc)*32)+lane)*16+e
// Lane l holds column n = nt*16 + (l&15); K-halves: group ko=(l>>4)*8:
//   e in [0,8)  -> k_local = kc*32 + ko + e
//   e in [8,16) -> k_local = kc*32 + ko + 16 + (e-8)
// Matches the 16-bit A/B 16x32 VGPR layout in the CDNA5 ISA (sec 7.12.2).
// ---------------------------------------------------------------------------
__global__ void sn_pack_kernel(const float* __restrict__ W,
                               const float* __restrict__ bvec,
                               const float* __restrict__ u,
                               int cin, int cout, int CinPad, int CoutPad,
                               _Float16* __restrict__ wfrag,
                               float* __restrict__ biasOut)
{
    __shared__ float t[96 * 9];   // K <= 864
    __shared__ float rr[96];
    __shared__ float red0, red1, sig;

    const int tid = threadIdx.x;
    const int K = cin * 9;

    if (tid == 0) { red0 = 0.0f; red1 = 0.0f; }
    __syncthreads();

    // t = Wm^T @ u   (Wm is cout x K, row-major)
    float lacc = 0.0f;
    for (int k = tid; k < K; k += blockDim.x) {
        float s = 0.0f;
        for (int n = 0; n < cout; ++n) s += W[n * K + k] * u[n];
        t[k] = s;
        lacc += s * s;
    }
    atomicAdd(&red0, lacc);
    __syncthreads();

    const float vscale = 1.0f / (sqrtf(red0) + 1e-12f);

    // r = Wm @ v ;  sigma = (r.r) / (||r|| + eps)
    lacc = 0.0f;
    for (int n = tid; n < cout; n += blockDim.x) {
        float s = 0.0f;
        for (int k = 0; k < K; ++k) s += W[n * K + k] * t[k];
        s *= vscale;
        rr[n] = s;
        lacc += s * s;
    }
    atomicAdd(&red1, lacc);
    __syncthreads();
    if (tid == 0) sig = red1 / (sqrtf(red1) + 1e-12f);
    __syncthreads();

    const float inv = 1.0f / sig;

    for (int n = tid; n < CoutPad; n += blockDim.x)
        biasOut[n] = (n < cout) ? bvec[n] : 0.0f;

    const int KC = CinPad >> 5;
    const int NT = CoutPad >> 4;
    const int total = NT * 9 * KC * 32 * 16;
    for (int idx = tid; idx < total; idx += blockDim.x) {
        int e    = idx & 15;
        int lane = (idx >> 4) & 31;
        int rest = idx >> 9;
        int kc   = rest % KC;  rest /= KC;
        int tap  = rest % 9;
        int nt   = rest / 9;

        int n  = nt * 16 + (lane & 15);
        int ko = (lane >> 4) * 8;
        int c  = kc * 32 + ko + (e >> 3) * 16 + (e & 7);

        float val = 0.0f;
        if (n < cout && c < cin) {
            int ky = tap / 3, kx = tap % 3;
            val = W[((n * cin + c) * 3 + ky) * 3 + kx] * inv;
        }
        wfrag[idx] = (_Float16)val;
    }
}

// ---------------------------------------------------------------------------
// Bilinear crop-resize of 3 sources (for/cur/back) into 9-channel fp16 patches,
// NHWC with channels zero-padded to 32. One thread per output pixel.
// Channel row written as 4x b128 vector stores (dst is 64B aligned).
// ---------------------------------------------------------------------------
__global__ void crop_kernel(const float* __restrict__ x,
                            const float* __restrict__ input_x,
                            const int* __restrict__ bboxes,
                            _Float16* __restrict__ act)
{
    int gid = blockIdx.x * blockDim.x + threadIdx.x;  // NPATCH * 64 * 64
    int p   = gid >> 12;
    int pix = gid & 4095;
    int oy  = pix >> 6, ox = pix & 63;
    int b   = p >> 2, tr = p & 3;

    float vals[9];

#pragma unroll
    for (int src = 0; src < 3; ++src) {
        const int* box = bboxes + (((b * 3 + src) * TRACK + tr) * 4);
        float h0 = (float)box[0], w0 = (float)box[1];
        float h1 = (float)box[2], w1 = (float)box[3];
        float ty = (float)oy * (1.0f / 63.0f);
        float tx = (float)ox * (1.0f / 63.0f);
        float ys = h0 + ty * (h1 - h0);
        float xs = w0 + tx * (w1 - w0);
        int y0 = (int)floorf(ys); y0 = min(max(y0, 0), H_IMG - 1);
        int x0 = (int)floorf(xs); x0 = min(max(x0, 0), W_IMG - 1);
        int y1 = min(y0 + 1, H_IMG - 1);
        int x1 = min(x0 + 1, W_IMG - 1);
        float fy = ys - (float)y0;
        float fx = xs - (float)x0;

        const float* img;
        if (src == 0)      img = input_x + (size_t)b * 6 * H_IMG * W_IMG;
        else if (src == 1) img = x + (size_t)b * 3 * H_IMG * W_IMG;
        else               img = input_x + (size_t)b * 6 * H_IMG * W_IMG
                                         + (size_t)3 * H_IMG * W_IMG;

#pragma unroll
        for (int c = 0; c < 3; ++c) {
            const float* ch = img + (size_t)c * H_IMG * W_IMG;
            float v00 = ch[y0 * W_IMG + x0], v01 = ch[y0 * W_IMG + x1];
            float v10 = ch[y1 * W_IMG + x0], v11 = ch[y1 * W_IMG + x1];
            float top = v00 * (1.0f - fx) + v01 * fx;
            float bot = v10 * (1.0f - fx) + v11 * fx;
            vals[src * 3 + c] = top * (1.0f - fy) + bot * fy;
        }
    }

    v8h o0 = {}, o1 = {}, z = {};
#pragma unroll
    for (int c = 0; c < 8; ++c) o0[c] = (_Float16)vals[c];
    o1[0] = (_Float16)vals[8];

    v8h* dst = (v8h*)(act + (size_t)gid * 32);
    dst[0] = o0;
    dst[1] = o1;
    dst[2] = z;
    dst[3] = z;
}

// ---------------------------------------------------------------------------
// Implicit-GEMM 3x3 conv using v_wmma_f32_16x16x32_f16.
// 128 threads = 4 waves per block; each wave owns one M-tile (16 output
// pixels), adjacent tiles share halo rows -> L0/L1 reuse. EXEC all-ones at
// every WMMA. N = NT x 16 output channels kept as live accumulators so each
// A fragment feeds NT WMMAs; K = 9 taps x KC x 32 channels.
// Wout is a power of two -> shift/mask instead of integer division.
// ---------------------------------------------------------------------------
template<int KC, int NT>
__global__ void conv_wmma_kernel(const _Float16* __restrict__ in,
                                 _Float16* __restrict__ out,
                                 const _Float16* __restrict__ wfrag,
                                 const float* __restrict__ bias,
                                 int Hin, int Win, int Hout, int Wout,
                                 int woutShift, int stride, int leaky)
{
    const int CinPad  = KC * 32;
    const int CoutPad = NT * 16;

    const int img  = blockIdx.y;
    const int wave = threadIdx.x >> 5;
    const int tile = blockIdx.x * 4 + wave;
    const int lane = threadIdx.x & 31;
    const int row  = lane & 15;
    const int hi   = lane >> 4;
    const int m    = tile * 16 + row;             // this lane's A-matrix row
    const int oy   = m >> woutShift;
    const int ox   = m & (Wout - 1);
    const int koff = hi * 8;                      // K sub-group per ISA A layout

    v8f zero = {};
    v8f acc[NT];
#pragma unroll
    for (int nt = 0; nt < NT; ++nt) acc[nt] = zero;

#pragma unroll
    for (int tap = 0; tap < 9; ++tap) {
        const int ky = tap / 3, kx = tap % 3;
        const int iy = oy * stride - 1 + ky;
        const int ix = ox * stride - 1 + kx;
        const bool valid = (iy >= 0) && (iy < Hin) && (ix >= 0) && (ix < Win);
        const _Float16* abase =
            in + (((size_t)img * Hin + iy) * Win + ix) * CinPad;

#pragma unroll
        for (int kc = 0; kc < KC; ++kc) {
            v8h g0 = {}, g1 = {};
            if (valid) {
                const v8h* pa = (const v8h*)(abase + kc * 32 + koff);
                __builtin_prefetch(pa, 0, 1);
                g0 = pa[0];   // K = kc*32 + koff + [0..7]
                g1 = pa[2];   // K = kc*32 + koff + 16 + [0..7]
            }
            v16h a;
#pragma unroll
            for (int i = 0; i < 8; ++i) { a[i] = g0[i]; a[i + 8] = g1[i]; }

#pragma unroll
            for (int nt = 0; nt < NT; ++nt) {
                const v8h* pb = (const v8h*)(wfrag +
                    ((((size_t)nt * 9 + tap) * KC + kc) * 32 + lane) * 16);
                v8h b0 = pb[0], b1 = pb[1];
                v16h bm;
#pragma unroll
                for (int i = 0; i < 8; ++i) { bm[i] = b0[i]; bm[i + 8] = b1[i]; }
                acc[nt] = __builtin_amdgcn_wmma_f32_16x16x32_f16(
                    false, a, false, bm, (short)0, acc[nt], false, false);
            }
        }
    }

    // Epilogue: D layout — lane l, VGPR r -> D[M = r + 8*(l>>4)][N = l&15]
#pragma unroll
    for (int nt = 0; nt < NT; ++nt) {
        const int n = nt * 16 + row;
        const float bn = bias[n];
#pragma unroll
        for (int r = 0; r < 8; ++r) {
            const int mr = tile * 16 + r + 8 * hi;
            float v = acc[nt][r] + bn;
            if (leaky) v = (v > 0.0f) ? v : 0.2f * v;
            out[((size_t)img * Hout * Wout + mr) * CoutPad + n] = (_Float16)v;
        }
    }
}

// ---------------------------------------------------------------------------
// Final reduction: mean over 8x8 spatial (channel 0) then mean over 4 tracks.
// ---------------------------------------------------------------------------
__global__ void reduce_kernel(const _Float16* __restrict__ act,
                              float* __restrict__ outp)
{
    __shared__ float s[NPATCH];
    const int p = threadIdx.x;  // 0..127
    float a = 0.0f;
    for (int q = 0; q < 64; ++q)
        a += (float)act[((size_t)p * 64 + q) * 32];
    s[p] = a;
    __syncthreads();
    if (p < BS) {
        float v = s[p * 4] + s[p * 4 + 1] + s[p * 4 + 2] + s[p * 4 + 3];
        outp[p] = v * (1.0f / (4.0f * 64.0f));
    }
}

// ---------------------------------------------------------------------------
// Host orchestration
// ---------------------------------------------------------------------------
extern "C" void kernel_launch(void* const* d_in, const int* in_sizes, int n_in,
                              void* d_out, int out_size, void* d_ws, size_t ws_size,
                              hipStream_t stream)
{
    (void)in_sizes; (void)n_in; (void)out_size; (void)ws_size;

    const float* x       = (const float*)d_in[0];
    const float* input_x = (const float*)d_in[2];
    const int*   bboxes  = (const int*)d_in[4];
    float*       outp    = (float*)d_out;

    static const int cin_[9]   = { 9, 16, 16, 32, 32, 64, 64, 96, 96 };
    static const int cout_[9]  = { 16, 16, 32, 32, 64, 64, 96, 96, 1 };
    static const int str_[9]   = { 1, 1, 2, 1, 2, 1, 2, 1, 1 };
    static const int hin_[9]   = { 64, 64, 64, 32, 32, 16, 16, 8, 8 };
    static const int hout_[9]  = { 64, 64, 32, 32, 16, 16, 8, 8, 8 };
    static const int hsh_[9]   = { 6, 6, 5, 5, 4, 4, 3, 3, 3 };  // log2(hout)
    static const int cinp_[9]  = { 32, 32, 32, 32, 32, 64, 64, 96, 96 };
    static const int coutp_[9] = { 32, 32, 32, 32, 64, 64, 96, 96, 32 };

    // Workspace carve-up (fp16 activations ping-pong + packed weights + bias)
    char* ws = (char*)d_ws;
    const size_t actElems = (size_t)NPATCH * 64 * 64 * 32;  // halves
    _Float16* actA = (_Float16*)ws;
    _Float16* actB = actA + actElems;
    char* cur = (char*)(actB + actElems);
    _Float16* wfrag[9];
    float*    biasp[9];
    for (int l = 0; l < 9; ++l) {
        size_t wn = (size_t)9 * cinp_[l] * coutp_[l];
        wfrag[l] = (_Float16*)cur; cur += wn * sizeof(_Float16);
        biasp[l] = (float*)cur;    cur += (size_t)coutp_[l] * sizeof(float);
    }

    // 1) Spectral-norm sigma + weight packing (one block per layer)
    for (int l = 0; l < 9; ++l) {
        const float* W = (const float*)d_in[5 + 3 * l];
        const float* b = (const float*)d_in[6 + 3 * l];
        const float* u = (const float*)d_in[7 + 3 * l];
        sn_pack_kernel<<<1, 256, 0, stream>>>(W, b, u, cin_[l], cout_[l],
                                              cinp_[l], coutp_[l],
                                              wfrag[l], biasp[l]);
    }

    // 2) Crop + bilinear resize into fp16 NHWC patches (C padded to 32)
    crop_kernel<<<(NPATCH * 64 * 64) / 256, 256, 0, stream>>>(x, input_x,
                                                              bboxes, actA);

    // 3) Conv stack via WMMA implicit GEMM, ping-pong buffers.
    //    4 waves per block, each wave one M-tile (tile counts all /4).
    const _Float16* inb = actA;
    _Float16*       oub = actB;
    for (int l = 0; l < 9; ++l) {
        const int tiles = hout_[l] * hout_[l] / 16;
        dim3 g(tiles / 4, NPATCH);
        const int leaky = (l < 8) ? 1 : 0;
        const int KC = cinp_[l] / 32;
        const int NT = coutp_[l] / 16;
        if (KC == 1 && NT == 2)
            conv_wmma_kernel<1, 2><<<g, 128, 0, stream>>>(inb, oub, wfrag[l], biasp[l],
                hin_[l], hin_[l], hout_[l], hout_[l], hsh_[l], str_[l], leaky);
        else if (KC == 1 && NT == 4)
            conv_wmma_kernel<1, 4><<<g, 128, 0, stream>>>(inb, oub, wfrag[l], biasp[l],
                hin_[l], hin_[l], hout_[l], hout_[l], hsh_[l], str_[l], leaky);
        else if (KC == 2 && NT == 4)
            conv_wmma_kernel<2, 4><<<g, 128, 0, stream>>>(inb, oub, wfrag[l], biasp[l],
                hin_[l], hin_[l], hout_[l], hout_[l], hsh_[l], str_[l], leaky);
        else if (KC == 2 && NT == 6)
            conv_wmma_kernel<2, 6><<<g, 128, 0, stream>>>(inb, oub, wfrag[l], biasp[l],
                hin_[l], hin_[l], hout_[l], hout_[l], hsh_[l], str_[l], leaky);
        else if (KC == 3 && NT == 6)
            conv_wmma_kernel<3, 6><<<g, 128, 0, stream>>>(inb, oub, wfrag[l], biasp[l],
                hin_[l], hin_[l], hout_[l], hout_[l], hsh_[l], str_[l], leaky);
        else
            conv_wmma_kernel<3, 2><<<g, 128, 0, stream>>>(inb, oub, wfrag[l], biasp[l],
                hin_[l], hin_[l], hout_[l], hout_[l], hsh_[l], str_[l], leaky);
        const _Float16* t = inb; inb = oub; oub = (_Float16*)t;
    }

    // 4) Spatial + track mean -> (BS, 1)
    reduce_kernel<<<1, NPATCH, 0, stream>>>(inb, outp);
}
